// Model1_23983097380894
// MI455X (gfx1250) — matrix operand order repaired and checked by reading, not compiled, not run on hardware
//
#include <hip/hip_runtime.h>
#include <hip/hip_bf16.h>
#include <math.h>

// ---------------------------------------------------------------- constants
#define CDIM 256      // n_components
#define DDIM 8192     // n_curves / time length
#define QLAG 32       // truncation_q
#define HID  4096
#define Y0   512
#define Y1   2048
#define WNH  64
#define TWOD 16384

// GEMM tiling (K of every GEMM is a multiple of 2*BK)
#define BM 64
#define BN 128
#define BK 32

typedef __attribute__((ext_vector_type(16))) __bf16 v16bf;
typedef __attribute__((ext_vector_type(8)))  __bf16 v8bf;
typedef __attribute__((ext_vector_type(4)))  __bf16 v4bf;
typedef __attribute__((ext_vector_type(8)))  float  v8f;
typedef __attribute__((ext_vector_type(4)))  unsigned int v4u;
typedef __attribute__((ext_vector_type(8)))  int    v8i;
typedef __attribute__((ext_vector_type(4)))  int    v4i;

#if __has_builtin(__builtin_amdgcn_tensor_load_to_lds) && __has_builtin(__builtin_amdgcn_s_wait_tensorcnt)
#define USE_TDM 1
#else
#define USE_TDM 0
#endif

__device__ __forceinline__ v4bf cvt4(float4 v) {
    v4bf r;
    r[0] = (__bf16)v.x; r[1] = (__bf16)v.y; r[2] = (__bf16)v.z; r[3] = (__bf16)v.w;
    return r;
}

__device__ __forceinline__ float gelu_exact(float x) {
    return 0.5f * x * (1.0f + erff(x * 0.70710678118654752f));
}

// ---------------------------------------------------------------- row means
__global__ __launch_bounds__(256) void mean_kernel(const float* __restrict__ X,
                                                   float* __restrict__ mu) {
    __shared__ float red[256];
    const int row = blockIdx.x;
    float s = 0.0f;
    for (int j = threadIdx.x; j < DDIM; j += 256)
        s += X[(size_t)row * DDIM + j];
    red[threadIdx.x] = s;
    __syncthreads();
    for (int off = 128; off > 0; off >>= 1) {
        if (threadIdx.x < off) red[threadIdx.x] += red[threadIdx.x + off];
        __syncthreads();
    }
    if (threadIdx.x == 0) mu[row] = red[0] * (1.0f / (float)DDIM);
}

// -------------------------------------------- weight network -> lag coeffs
__global__ __launch_bounds__(128) void coef_kernel(const float* __restrict__ w1,
                                                   const float* __restrict__ b1,
                                                   const float* __restrict__ w2,
                                                   const float* __restrict__ b2,
                                                   float* __restrict__ coef) {
    __shared__ float w[2 * QLAG + 1];
    const int t = threadIdx.x;
    if (t < 2 * QLAG + 1) {
        float lag = (float)(t - QLAG);
        float acc = b2[0];
        for (int h = 0; h < WNH; ++h) {
            float z = lag * w1[h] + b1[h];
            acc += gelu_exact(z) * w2[h];
        }
        w[t] = acc;
    }
    __syncthreads();
    if (t <= QLAG) {
        float c;
        if (t == 0) c = w[QLAG] * (1.0f / (float)DDIM);
        else        c = (w[QLAG - t] + w[QLAG + t]) / (float)(DDIM - t);
        coef[t] = c;
    }
}

// --------- demean + write Xo^T + FIR-filtered Y; X row tile staged via TDM
__global__ __launch_bounds__(256) void prep_kernel(const float* __restrict__ X,
                                                   const float* __restrict__ mu,
                                                   const float* __restrict__ coef,
                                                   float* __restrict__ XoT,
                                                   float* __restrict__ Y) {
    __shared__ __align__(16) float xs[256 + QLAG];   // raw X values (0 past row end)
    const int row  = blockIdx.y;
    const int base = blockIdx.x * 256;
    const int t    = threadIdx.x;
    const float m  = mu[row];
#if USE_TDM
    if (t < 32) {   // wave 0 issues one Tensor Data Mover descriptor
        const unsigned long long ga =
            (unsigned long long)(const void*)(X + (size_t)row * DDIM + base);
        const unsigned int lds = (unsigned int)(unsigned long long)(const void*)&xs[0];
        const unsigned int t0  = (unsigned int)(DDIM - base); // OOB tail reads -> 0
        // D# group 0: count=1 | lds_addr | global_addr | type=2
        v4u g0 = { 1u,
                   lds,
                   (unsigned int)ga,
                   (unsigned int)((ga >> 32) & 0x1FFFFFFu) | 0x80000000u };
        // D# group 1: data_size=4B, tensor_dim0=t0, tensor_dim1=1,
        //             tile_dim0=288, tile_dim1=1, dim0_stride=DDIM
        v8i g1 = { (int)0x00020000u,
                   (int)((t0 & 0xFFFFu) << 16),
                   (int)(((t0 >> 16) & 0xFFFFu) | (1u << 16)),
                   (int)((unsigned)(256 + QLAG) << 16),
                   1,
                   (int)DDIM,
                   0, 0 };
        v4i gz4 = { 0, 0, 0, 0 };
        v8i gz8 = { 0, 0, 0, 0, 0, 0, 0, 0 };
        __builtin_amdgcn_tensor_load_to_lds(g0, g1, gz4, gz4, gz8, 0);
        __builtin_amdgcn_s_wait_tensorcnt(0);
    }
    __syncthreads();
#else
    for (int j = t; j < 256 + QLAG; j += 256) {
        int col = base + j;
        xs[j] = (col < DDIM) ? X[(size_t)row * DDIM + col] : 0.0f;
    }
    __syncthreads();
#endif
    const int col = base + t;
    XoT[(size_t)col * CDIM + row] = xs[t] - m;
    // y = sum_k c_k * (x[col+k] - m), with Xo == 0 past row end:
    //   raw FIR (OOB x are 0) minus m * (sum of c_k over in-range taps)
    float acc = 0.0f, cs = 0.0f;
#pragma unroll
    for (int k = 0; k <= QLAG; ++k) {
        acc += coef[k] * xs[t + k];
        if (col + k < DDIM) cs += coef[k];
    }
    Y[(size_t)row * DDIM + col] = acc - m * cs;
}

// ------------------------------------ combined[:, :D] = X  (ldc = 2D)
__global__ __launch_bounds__(256) void copyx_kernel(const float* __restrict__ X,
                                                    float* __restrict__ comb) {
    const int idx = blockIdx.x * 256 + threadIdx.x;   // float4 index
    const int i  = idx / (DDIM / 4);
    const int j4 = idx % (DDIM / 4);
    const float4 v = *(const float4*)&X[(size_t)i * DDIM + j4 * 4];
    *(float4*)&comb[(size_t)i * TWOD + j4 * 4] = v;
}

// ---------------------------------------------------------------- WMMA GEMM
// C[M,N] (f32) = act( A[M,K] * B[K,N] + bias ), A/B f32 in memory, bf16 WMMA.
// Software-pipelined: register double-buffer over global loads, ping-pong
// LDS buffers, one barrier per K-step.  Requires K % (2*BK) == 0.
// EPI: 0 = plain store, 1 = bias + exact GELU.
template <int EPI>
__global__ __launch_bounds__(256) void gemm_bf16wmma(
        const float* __restrict__ A, int lda,
        const float* __restrict__ B, int ldb,
        const float* __restrict__ bias,
        float* __restrict__ Cout, int ldc,
        int M, int N, int K) {
    __shared__ __align__(16) __bf16 As[2][BM][BK];   // row-major (m, k)
    __shared__ __align__(16) __bf16 Bs[2][BN][BK];   // transposed (n, k)
    constexpr int ASTR = BM * BK;   // halves per A buffer (4 KB)
    constexpr int BSTR = BN * BK;   // halves per B buffer (8 KB)

    const int tid  = threadIdx.x;
    const int lane = tid & 31;
    const int wave = tid >> 5;        // 0..7
    const int wm   = wave >> 2;       // 0..1 -> 32-row strip
    const int wn   = wave & 3;        // 0..3 -> 32-col strip
    const int lr   = lane & 15;
    const int hi   = lane >> 4;
    const int rowBase = blockIdx.y * BM;
    const int colBase = blockIdx.x * BN;

    // ---- hoisted staging pointers (advance by constant strides per K-step)
    const float* pA[2]; __bf16* sA[2];
#pragma unroll
    for (int i = 0; i < 2; ++i) {
        int f = tid + i * 256, r = f >> 3, kq = f & 7;
        pA[i] = A + (size_t)(rowBase + r) * lda + kq * 4;
        sA[i] = &As[0][r][kq * 4];
    }
    const float* pB[4]; __bf16* sB[4];
#pragma unroll
    for (int i = 0; i < 4; ++i) {
        int u = tid + i * 256, n = u & 127, kq = u >> 7;
        pB[i] = B + (size_t)(kq * 4) * ldb + colBase + n;
        sB[i] = &Bs[0][n][kq * 4];
    }
    // ---- uniform-branch prefetch pointers, 2 tiles ahead
    const float* pfB = B + (size_t)(2 * BK) * ldb + colBase
                         + (size_t)((tid & 63) >> 1) * ldb + (tid & 1) * 64;
    const float* pfA = A + (size_t)(rowBase + (tid & 63)) * lda + 2 * BK;

    // ---- hoisted fragment LDS pointers (CDNA5 ISA 7.12.2 wave32 layouts)
    const __bf16* fA0[2]; const __bf16* fA1[2];
#pragma unroll
    for (int mf = 0; mf < 2; ++mf) {
        int r = wm * 32 + mf * 16 + lr;
        fA0[mf] = &As[0][r][hi * 8];
        fA1[mf] = &As[0][r][16 + hi * 8];
    }
    const __bf16* fB0[2]; const __bf16* fB1[2];
#pragma unroll
    for (int nf = 0; nf < 2; ++nf) {
        int n = wn * 32 + nf * 16 + lr;
        fB0[nf] = &Bs[0][n][hi * 8];
        fB1[nf] = &Bs[0][n][16 + hi * 8];
    }

    const v8f vzero = {0.f, 0.f, 0.f, 0.f, 0.f, 0.f, 0.f, 0.f};
    v8f acc[2][2] = {{vzero, vzero}, {vzero, vzero}};
    float4 ra[2];        // register double-buffer for A tile
    float4 rb[4];        // register double-buffer for B tile

    auto gload = [&]() {                 // issue next tile's global loads
#pragma unroll
        for (int i = 0; i < 2; ++i) { ra[i] = *(const float4*)pA[i]; pA[i] += BK; }
#pragma unroll
        for (int i = 0; i < 4; ++i) {
            rb[i].x = pB[i][0];
            rb[i].y = pB[i][(size_t)ldb];
            rb[i].z = pB[i][(size_t)2 * ldb];
            rb[i].w = pB[i][(size_t)3 * ldb];
            pB[i] += (size_t)BK * ldb;
        }
    };
    auto stage = [&](int buf) {          // regs -> bf16 -> LDS[buf]
#pragma unroll
        for (int i = 0; i < 2; ++i) *(v4bf*)(sA[i] + buf * ASTR) = cvt4(ra[i]);
#pragma unroll
        for (int i = 0; i < 4; ++i) *(v4bf*)(sB[i] + buf * BSTR) = cvt4(rb[i]);
    };
    auto compute = [&](int buf) {        // 4 WMMAs from LDS[buf]
        union ABu { v16bf v; v8bf h[2]; };
        v16bf af[2], bfr[2];
#pragma unroll
        for (int mf = 0; mf < 2; ++mf) {
            ABu u;
            u.h[0] = *(const v8bf*)(fA0[mf] + buf * ASTR);
            u.h[1] = *(const v8bf*)(fA1[mf] + buf * ASTR);
            af[mf] = u.v;
        }
#pragma unroll
        for (int nf = 0; nf < 2; ++nf) {
            ABu u;
            u.h[0] = *(const v8bf*)(fB0[nf] + buf * BSTR);
            u.h[1] = *(const v8bf*)(fB1[nf] + buf * BSTR);
            bfr[nf] = u.v;
        }
#pragma unroll
        for (int mf = 0; mf < 2; ++mf)
#pragma unroll
            for (int nf = 0; nf < 2; ++nf)
                acc[mf][nf] = __builtin_amdgcn_wmma_f32_16x16x32_bf16(
                    false, af[mf], false, bfr[nf],
                    (short)0, acc[mf][nf], false, false);
    };

    // ---- software pipeline (K/BK is even for every GEMM in this pipeline)
    gload();                     // tile 0
    stage(0);
    for (int k0 = 0; k0 < K; k0 += 2 * BK) {
        __syncthreads();                         // LDS[0] ready for all waves
        if (k0 + 2 * BK < K) {                   // uniform branch
            __builtin_prefetch(pfB, 0, 1);       // global_prefetch_b8
            __builtin_prefetch(pfA, 0, 1);
        }
        pfB += (size_t)(2 * BK) * ldb;
        pfA += 2 * BK;

        gload();                 // tile k0+BK (always in range: even count)
        compute(0);
        stage(1);
        __syncthreads();                         // LDS[1] ready for all waves
        if (k0 + 2 * BK < K) gload();            // tile k0+2BK
        compute(1);
        stage(0);                // stale regs on final pass: never read
    }

    // ---- epilogue: VGPR r holds row (r + 8*hi), lane lr = col
#pragma unroll
    for (int mf = 0; mf < 2; ++mf) {
#pragma unroll
        for (int nf = 0; nf < 2; ++nf) {
            int rbase = rowBase + wm * 32 + mf * 16 + hi * 8;
            int col   = colBase + wn * 32 + nf * 16 + lr;
            float bv = 0.0f;
            if (EPI == 1) bv = bias[col];
#pragma unroll
            for (int r = 0; r < 8; ++r) {
                float x = acc[mf][nf][r];
                if (EPI == 1) x = gelu_exact(x + bv);
                Cout[(size_t)(rbase + r) * ldc + col] = x;
            }
        }
    }
}

// ---------------------------------------------------------------- launcher
extern "C" void kernel_launch(void* const* d_in, const int* in_sizes, int n_in,
                              void* d_out, int out_size, void* d_ws, size_t ws_size,
                              hipStream_t stream) {
    (void)in_sizes; (void)n_in; (void)out_size; (void)ws_size;

    const float* X     = (const float*)d_in[0];
    const float* wn_w1 = (const float*)d_in[1];
    const float* wn_b1 = (const float*)d_in[2];
    const float* wn_w2 = (const float*)d_in[3];
    const float* wn_b2 = (const float*)d_in[4];
    const float* fc1_w = (const float*)d_in[5];
    const float* fc1_b = (const float*)d_in[6];
    const float* fc2_w = (const float*)d_in[7];
    const float* fc2_b = (const float*)d_in[8];
    const float* proj  = (const float*)d_in[9];
    float* out = (float*)d_out;

    // workspace layout (floats)
    float* ws = (float*)d_ws;
    float* mu   = ws;                                  // 256
    float* coef = ws + 256;                            // 256 (33 used)
    float* Yf   = ws + 512;                            // C*D
    float* XoT  = Yf   + (size_t)CDIM * DDIM;          // D*C
    float* cov  = XoT  + (size_t)DDIM * CDIM;          // C*C
    float* comb = cov  + (size_t)CDIM * CDIM;          // C*2D
    float* h1   = comb + (size_t)CDIM * TWOD;          // C*HID
    float* h2   = h1   + (size_t)CDIM * HID;           // C*Y1

    mean_kernel<<<CDIM, 256, 0, stream>>>(X, mu);
    coef_kernel<<<1, 128, 0, stream>>>(wn_w1, wn_b1, wn_w2, wn_b2, coef);
    prep_kernel<<<dim3(DDIM / 256, CDIM), 256, 0, stream>>>(X, mu, coef, XoT, Yf);
    copyx_kernel<<<(CDIM * DDIM / 4) / 256, 256, 0, stream>>>(X, comb);

    // cov = Y @ Xo^T                  (256 x 256, K=8192)
    gemm_bf16wmma<0><<<dim3(CDIM / BN, CDIM / BM), 256, 0, stream>>>(
        Yf, DDIM, XoT, CDIM, nullptr, cov, CDIM, CDIM, CDIM, DDIM);
    // combined[:, D:] = cov @ X       (256 x 8192, K=256)
    gemm_bf16wmma<0><<<dim3(DDIM / BN, CDIM / BM), 256, 0, stream>>>(
        cov, CDIM, X, DDIM, nullptr, comb + DDIM, TWOD, CDIM, DDIM, CDIM);
    // h1 = gelu(combined @ fc1_w + b1)   (256 x 4096, K=16384)
    gemm_bf16wmma<1><<<dim3(HID / BN, CDIM / BM), 256, 0, stream>>>(
        comb, TWOD, fc1_w, HID, fc1_b, h1, HID, CDIM, HID, TWOD);
    // h2 = gelu(h1 @ fc2_w + b2)         (256 x 2048, K=4096)
    gemm_bf16wmma<1><<<dim3(Y1 / BN, CDIM / BM), 256, 0, stream>>>(
        h1, HID, fc2_w, Y1, fc2_b, h2, Y1, CDIM, Y1, HID);
    // out = proj @ h2                    (512 x 2048, K=256)
    gemm_bf16wmma<0><<<dim3(Y1 / BN, Y0 / BM), 256, 0, stream>>>(
        proj, CDIM, h2, Y1, nullptr, out, Y1, Y0, Y1, CDIM);
}
